// Model_27891517620704
// MI455X (gfx1250) — compile-verified, matrix operand order checked by reading, not run on hardware
//
#include <hip/hip_runtime.h>
#include <hip/hip_bf16.h>

// ---------------------------------------------------------------------------
// GAT(3-head,48) -> ELU -> GAT(1-head,128) -> log_softmax -> GCN -> ReLU -> Linear
// MI455X strategy: pipeline is edge-scatter (HBM/L2-atomic) bound (~2.5 GB /
// call -> ~110us floor @ 23.3 TB/s). Dense GEMMs (~6.3 GFLOP) run on
// v_wmma_f32_16x16x32_f16 with double-buffered LDS f16 tiles. B is stored
// transposed in LDS so both fragments load as two ds_load_b128 per operand
// (ISA 7.12.2: k = (h&8 ? h+8 : h) + (lane>=16 ? 8 : 0) -> two contiguous
// 8-half runs). Global tile loads are unconditional float4 with cndmask
// zeroing (no exec-divergent scalar loads).
// ---------------------------------------------------------------------------

typedef __attribute__((ext_vector_type(16))) _Float16 v16h;
typedef __attribute__((ext_vector_type(8)))  _Float16 v8h;
typedef __attribute__((ext_vector_type(4)))  _Float16 v4h;
typedef __attribute__((ext_vector_type(8)))  float    v8f;

#define TK 32          // WMMA K step
#define LDS_PITCH 40   // halfs per row: 80B, keeps 16B alignment + bank skew
#define GEMM_THREADS 128

union frag16 { v16h v; v8h h[2]; };

// cooperative tile load: A[16 x 32] row-major f16, B[32 x 64] stored TRANSPOSED
__device__ __forceinline__
void load_tile(const float* __restrict__ A, const float* __restrict__ B,
               int M, int K, int Nout, int m0, int nb0, int kk, int tid,
               _Float16 (*Ah)[LDS_PITCH], _Float16 (*BhT)[LDS_PITCH])
{
    // ---- A: 16x32 floats = 128 float4, one per thread (coalesced b128) ----
    {
        int r  = tid >> 3;
        int k4 = (tid & 7) << 2;
        int gm = m0 + r, gk = kk + k4;
        bool valid = (gm < M) && (gk < K);            // K,Nout multiples of 4
        long long off = (long long)min(gm, M - 1) * K + min(gk, K - 4);
        float4 va = *(const float4*)(A + off);
        if (!valid) va = make_float4(0.f, 0.f, 0.f, 0.f);
        v4h h = { (_Float16)va.x, (_Float16)va.y, (_Float16)va.z, (_Float16)va.w };
        *(v4h*)&Ah[r][k4] = h;
        // prefetch tile after next (global_prefetch_b8, speculative-safe)
        __builtin_prefetch(A + off + 2 * TK, 0, 1);
    }
    // ---- B: 32x64 floats = 512 float4, four per thread; transposed store ---
    #pragma unroll
    for (int u = 0; u < 4; ++u) {
        int idx = tid + u * GEMM_THREADS;
        int k   = idx >> 4;
        int n4  = (idx & 15) << 2;
        int gk = kk + k, gn = nb0 + n4;
        bool valid = (gk < K) && (gn + 3 < Nout);
        long long off = (long long)min(gk, K - 1) * Nout + min(gn, max(Nout - 4, 0));
        float4 vb = *(const float4*)(B + off);
        if (!valid) vb = make_float4(0.f, 0.f, 0.f, 0.f);
        BhT[n4 + 0][k] = (_Float16)vb.x;
        BhT[n4 + 1][k] = (_Float16)vb.y;
        BhT[n4 + 2][k] = (_Float16)vb.z;
        BhT[n4 + 3][k] = (_Float16)vb.w;
    }
}

// ---------------- WMMA GEMM: C[M x Nout] = A[M x K] @ B[K x Nout] (+bias) ---
__global__ __launch_bounds__(GEMM_THREADS)
void k_gemm_wmma(const float* __restrict__ A, const float* __restrict__ B,
                 const float* __restrict__ bias, float* __restrict__ C,
                 int M, int K, int Nout)
{
    __shared__ __align__(16) _Float16 Ah [2][16][LDS_PITCH];  // 2 x 1.25 KB
    __shared__ __align__(16) _Float16 BhT[2][64][LDS_PITCH];  // 2 x 5 KB

    const int tid  = threadIdx.x;
    const int lane = tid & 31;
    const int wave = tid >> 5;
    const int m0   = blockIdx.x * 16;
    const int nb0  = blockIdx.y * 64;
    const int n0   = nb0 + wave * 16;

    v8f acc = {0.f, 0.f, 0.f, 0.f, 0.f, 0.f, 0.f, 0.f};

    const int ml = lane & 15;
    const int kb = (lane >> 4) ? 8 : 0;   // lanes 16-31 hold K halves +8 / +24
    const int brow = wave * 16 + ml;

    load_tile(A, B, M, K, Nout, m0, nb0, 0, tid, Ah[0], BhT[0]);
    __syncthreads();

    int p = 0;
    for (int kk = 0; kk < K; kk += TK) {
        if (kk + TK < K)   // prefetch next K-tile into the other stage
            load_tile(A, B, M, K, Nout, m0, nb0, kk + TK, tid, Ah[p ^ 1], BhT[p ^ 1]);

        // fragments: two contiguous 16B runs each (ds_load_b128 x2 per operand)
        frag16 af, bf;
        af.h[0] = *(const v8h*)&Ah[p][ml][kb];
        af.h[1] = *(const v8h*)&Ah[p][ml][16 + kb];
        bf.h[0] = *(const v8h*)&BhT[p][brow][kb];
        bf.h[1] = *(const v8h*)&BhT[p][brow][16 + kb];

        acc = __builtin_amdgcn_wmma_f32_16x16x32_f16(false, af.v, false, bf.v,
                                                     (short)0, acc, false, false);
        __syncthreads();
        p ^= 1;
    }

    // store D (VGPR r: lanes 0-15 -> M=r, lanes 16-31 -> M=r+8; N = lane&15)
    const int col   = n0 + (lane & 15);
    const int rbase = (lane >> 4) << 3;
    if (col < Nout) {
        #pragma unroll
        for (int r = 0; r < 8; ++r) {
            int row = m0 + r + rbase;
            if (row < M) {
                float v = acc[r];
                if (bias) v += bias[col];
                C[(long long)row * Nout + col] = v;
            }
        }
    }
}

// ---------------- helpers ---------------------------------------------------
__device__ inline void edge_sd(const int* __restrict__ ei, long long E,
                               long long e, int& s, int& d)
{
    if (e < E) { s = ei[e]; d = ei[E + e]; }
    else       { s = d = (int)(e - E); }   // self loop
}

__device__ inline void atomicMaxFloat(float* addr, float val)
{
    if (val >= 0.f) atomicMax((int*)addr, __float_as_int(val));
    else            atomicMin((unsigned int*)addr, __float_as_uint(val));
}

__global__ void k_fill(float* p, long long n, float v)
{
    long long t = (long long)blockIdx.x * blockDim.x + threadIdx.x;
    if (t < n) p[t] = v;
}

// per-node attention scores: es/ed[n,h] = dot(h[n,h,:], a_src/dst[h,:])
__global__ void k_att_scores(const float* __restrict__ hraw,
                             const float* __restrict__ asrc,
                             const float* __restrict__ adst,
                             float* __restrict__ es, float* __restrict__ ed,
                             int N, int H, int C)
{
    long long t = (long long)blockIdx.x * blockDim.x + threadIdx.x;
    if (t >= (long long)N * H) return;
    int n = (int)(t / H), h = (int)(t % H);
    const float* row = hraw + (long long)n * H * C + h * C;
    float s1 = 0.f, s2 = 0.f;
    for (int c = 0; c < C; ++c) {
        float v = row[c];
        s1 += v * asrc[h * C + c];
        s2 += v * adst[h * C + c];
    }
    es[t] = s1; ed[t] = s2;
}

// edge pass 1: segment max of leaky_relu(es[s]+ed[d]) over dst
__global__ void k_edge_max(const int* __restrict__ ei, long long E, long long Etot,
                           const float* __restrict__ es, const float* __restrict__ ed,
                           float* __restrict__ m, int H)
{
    long long t = (long long)blockIdx.x * blockDim.x + threadIdx.x;
    if (t >= Etot * H) return;
    long long e = t / H; int h = (int)(t % H);
    int s, d; edge_sd(ei, E, e, s, d);
    float v = es[(long long)s * H + h] + ed[(long long)d * H + h];
    v = (v >= 0.f) ? v : 0.2f * v;
    atomicMaxFloat(&m[(long long)d * H + h], v);
}

// edge pass 2: w = exp(e - m[d]); store w, accumulate denominator
__global__ void k_edge_w(const int* __restrict__ ei, long long E, long long Etot,
                         const float* __restrict__ es, const float* __restrict__ ed,
                         const float* __restrict__ m, float* __restrict__ den,
                         float* __restrict__ wbuf, int H)
{
    long long t = (long long)blockIdx.x * blockDim.x + threadIdx.x;
    if (t >= Etot * H) return;
    long long e = t / H; int h = (int)(t % H);
    int s, d; edge_sd(ei, E, e, s, d);
    float v = es[(long long)s * H + h] + ed[(long long)d * H + h];
    v = (v >= 0.f) ? v : 0.2f * v;
    float w = __expf(v - m[(long long)d * H + h]);
    wbuf[t] = w;
    atomicAdd(&den[(long long)d * H + h], w);
}

// edge pass 3: num[d,:] += w * h[s,:]  (float4 gather + 4 f32 atomics)
__global__ void k_edge_scatter_gat(const int* __restrict__ ei, long long E, long long Etot,
                                   const float* __restrict__ wbuf,
                                   const float* __restrict__ hsrc,
                                   float* __restrict__ num, int H, int C)
{
    const int HC = H * C;
    const long long chunks = HC >> 2;
    long long t = (long long)blockIdx.x * blockDim.x + threadIdx.x;
    if (t >= Etot * chunks) return;
    long long e = t / chunks; int q = (int)(t % chunks);
    int s, d; edge_sd(ei, E, e, s, d);
    int j = q << 2;
    int h = j / C;
    float wv = wbuf[e * H + h];
    const float4 hv = *(const float4*)(hsrc + (long long)s * HC + j);
    float* p = num + (long long)d * HC + j;
    atomicAdd(p + 0, wv * hv.x);
    atomicAdd(p + 1, wv * hv.y);
    atomicAdd(p + 2, wv * hv.z);
    atomicAdd(p + 3, wv * hv.w);
}

// normalize + bias (+ optional ELU)
__global__ void k_gat_finalize(const float* __restrict__ num, const float* __restrict__ den,
                               const float* __restrict__ b, float* __restrict__ out,
                               int N, int H, int C, int do_elu)
{
    const int HC = H * C;
    long long t = (long long)blockIdx.x * blockDim.x + threadIdx.x;
    if (t >= (long long)N * HC) return;
    int n = (int)(t / HC), j = (int)(t % HC);
    int h = j / C;
    float v = num[t] / den[(long long)n * H + h] + b[j];
    if (do_elu) v = (v > 0.f) ? v : (__expf(v) - 1.f);
    out[t] = v;
}

// in-place log_softmax over C=128 per node; one wave32 per node
__global__ __launch_bounds__(128)
void k_logsoftmax(float* __restrict__ v, int N, int C)
{
    int node = (int)(((long long)blockIdx.x * blockDim.x + threadIdx.x) >> 5);
    int lane = threadIdx.x & 31;
    if (node >= N) return;
    float* row = v + (long long)node * C;
    float mx = -INFINITY;
    for (int j = lane; j < C; j += 32) mx = fmaxf(mx, row[j]);
    for (int o = 16; o; o >>= 1) mx = fmaxf(mx, __shfl_xor(mx, o, 32));
    float s = 0.f;
    for (int j = lane; j < C; j += 32) s += __expf(row[j] - mx);
    for (int o = 16; o; o >>= 1) s += __shfl_xor(s, o, 32);
    float lg = mx + __logf(s);
    for (int j = lane; j < C; j += 32) row[j] -= lg;
}

// GCN degree / inv-sqrt / scatter
__global__ void k_deg(const int* __restrict__ ei, long long E, long long Etot,
                      float* __restrict__ deg)
{
    long long e = (long long)blockIdx.x * blockDim.x + threadIdx.x;
    if (e >= Etot) return;
    int s, d; edge_sd(ei, E, e, s, d);
    atomicAdd(&deg[d], 1.f);
}

__global__ void k_dinv(const float* __restrict__ deg, float* __restrict__ dinv, int N)
{
    int n = blockIdx.x * blockDim.x + threadIdx.x;
    if (n < N) dinv[n] = rsqrtf(fmaxf(deg[n], 1.f));
}

__global__ void k_edge_scatter_gcn(const int* __restrict__ ei, long long E, long long Etot,
                                   const float* __restrict__ dinv,
                                   const float* __restrict__ hsrc,
                                   float* __restrict__ agg, int C)
{
    const long long chunks = C >> 2;
    long long t = (long long)blockIdx.x * blockDim.x + threadIdx.x;
    if (t >= Etot * chunks) return;
    long long e = t / chunks; int q = (int)(t % chunks);
    int s, d; edge_sd(ei, E, e, s, d);
    float nm = dinv[s] * dinv[d];
    int j = q << 2;
    const float4 hv = *(const float4*)(hsrc + (long long)s * C + j);
    float* p = agg + (long long)d * C + j;
    atomicAdd(p + 0, nm * hv.x);
    atomicAdd(p + 1, nm * hv.y);
    atomicAdd(p + 2, nm * hv.z);
    atomicAdd(p + 3, nm * hv.w);
}

__global__ void k_relu_bias(float* __restrict__ x, const float* __restrict__ b,
                            long long total, int C)
{
    long long t = (long long)blockIdx.x * blockDim.x + threadIdx.x;
    if (t >= total) return;
    float v = x[t] + b[t % C];
    x[t] = (v > 0.f) ? v : 0.f;
}

// ---------------------------------------------------------------------------
static inline unsigned gsz(long long n, int b) { return (unsigned)((n + b - 1) / b); }

extern "C" void kernel_launch(void* const* d_in, const int* in_sizes, int n_in,
                              void* d_out, int out_size, void* d_ws, size_t ws_size,
                              hipStream_t stream)
{
    const float* x      = (const float*)d_in[0];
    const int*   ei     = (const int*)  d_in[1];
    const float* W1     = (const float*)d_in[2];
    const float* a_src1 = (const float*)d_in[3];
    const float* a_dst1 = (const float*)d_in[4];
    const float* b1     = (const float*)d_in[5];
    const float* W2     = (const float*)d_in[6];
    const float* a_src2 = (const float*)d_in[7];
    const float* a_dst2 = (const float*)d_in[8];
    const float* b2     = (const float*)d_in[9];
    const float* Wg     = (const float*)d_in[10];
    const float* bg     = (const float*)d_in[11];
    const float* Wl     = (const float*)d_in[12];
    const float* bl     = (const float*)d_in[13];
    float* out = (float*)d_out;

    const int       IN_C = 512;
    const int       N    = in_sizes[0] / IN_C;          // 50000
    const long long E    = in_sizes[1] / 2;             // 800000
    const long long Etot = E + N;                       // + self loops

    // workspace layout (floats); heavy reuse keeps it ~110 MB (L2-resident)
    float* ws  = (float*)d_ws;
    size_t off = 0;
    auto take = [&](size_t n) { float* p = ws + off; off += n; return p; };
    float* h1   = take((size_t)N * 48);     // x@W1, later overwritten by ELU output f1
    float* num1 = take((size_t)N * 48);
    float* es1  = take((size_t)N * 3);
    float* ed1  = take((size_t)N * 3);
    float* m1   = take((size_t)N * 3);
    float* den1 = take((size_t)N * 3);
    float* wE   = take((size_t)Etot * 3);   // reused (H=1) for layer 2
    float* h2   = take((size_t)N * 128);    // f1@W2, later f2 (log_softmax output)
    float* num2 = take((size_t)N * 128);    // later reused as hg = f2@Wg
    float* es2  = take((size_t)N);
    float* ed2  = take((size_t)N);
    float* m2   = take((size_t)N);
    float* den2 = take((size_t)N);
    float* deg  = take((size_t)N);
    float* dinv = take((size_t)N);
    float* agg  = take((size_t)N * 128);    // GCN aggregate, relu'd in place -> f3

    const dim3 B256(256), B128(128);

    // ---------------- GAT layer 1 (H=3, C=16) ----------------
    {   // h1 = x @ W1   [N x 512] @ [512 x 48]
        dim3 g((N + 15) / 16, (48 + 63) / 64);
        k_gemm_wmma<<<g, B128, 0, stream>>>(x, W1, nullptr, h1, N, IN_C, 48);
    }
    k_att_scores<<<gsz((long long)N * 3, 256), B256, 0, stream>>>(h1, a_src1, a_dst1, es1, ed1, N, 3, 16);
    k_fill<<<gsz((long long)N * 3, 256), B256, 0, stream>>>(m1,   (long long)N * 3, -INFINITY);
    k_fill<<<gsz((long long)N * 3, 256), B256, 0, stream>>>(den1, (long long)N * 3, 0.f);
    k_fill<<<gsz((long long)N * 48, 256), B256, 0, stream>>>(num1, (long long)N * 48, 0.f);
    k_edge_max<<<gsz(Etot * 3, 256), B256, 0, stream>>>(ei, E, Etot, es1, ed1, m1, 3);
    k_edge_w  <<<gsz(Etot * 3, 256), B256, 0, stream>>>(ei, E, Etot, es1, ed1, m1, den1, wE, 3);
    k_edge_scatter_gat<<<gsz(Etot * 12, 256), B256, 0, stream>>>(ei, E, Etot, wE, h1, num1, 3, 16);
    k_gat_finalize<<<gsz((long long)N * 48, 256), B256, 0, stream>>>(num1, den1, b1, h1, N, 3, 16, /*elu*/1);

    // ---------------- GAT layer 2 (H=1, C=128) ----------------
    {   // h2 = f1 @ W2   [N x 48] @ [48 x 128]  (K zero-padded to 64 in LDS)
        dim3 g((N + 15) / 16, (128 + 63) / 64);
        k_gemm_wmma<<<g, B128, 0, stream>>>(h1, W2, nullptr, h2, N, 48, 128);
    }
    k_att_scores<<<gsz(N, 256), B256, 0, stream>>>(h2, a_src2, a_dst2, es2, ed2, N, 1, 128);
    k_fill<<<gsz(N, 256), B256, 0, stream>>>(m2,   N, -INFINITY);
    k_fill<<<gsz(N, 256), B256, 0, stream>>>(den2, N, 0.f);
    k_fill<<<gsz((long long)N * 128, 256), B256, 0, stream>>>(num2, (long long)N * 128, 0.f);
    k_edge_max<<<gsz(Etot, 256), B256, 0, stream>>>(ei, E, Etot, es2, ed2, m2, 1);
    k_edge_w  <<<gsz(Etot, 256), B256, 0, stream>>>(ei, E, Etot, es2, ed2, m2, den2, wE, 1);
    k_edge_scatter_gat<<<gsz(Etot * 32, 256), B256, 0, stream>>>(ei, E, Etot, wE, h2, num2, 1, 128);
    k_gat_finalize<<<gsz((long long)N * 128, 256), B256, 0, stream>>>(num2, den2, b2, h2, N, 1, 128, /*elu*/0);
    k_logsoftmax<<<gsz((long long)N * 32, 128), B128, 0, stream>>>(h2, N, 128);

    // ---------------- GCN + ReLU ----------------
    k_fill<<<gsz(N, 256), B256, 0, stream>>>(deg, N, 0.f);
    k_deg<<<gsz(Etot, 256), B256, 0, stream>>>(ei, E, Etot, deg);
    k_dinv<<<gsz(N, 256), B256, 0, stream>>>(deg, dinv, N);
    {   // hg = f2 @ Wg (into num2, free now)
        dim3 g((N + 15) / 16, (128 + 63) / 64);
        k_gemm_wmma<<<g, B128, 0, stream>>>(h2, Wg, nullptr, num2, N, 128, 128);
    }
    k_fill<<<gsz((long long)N * 128, 256), B256, 0, stream>>>(agg, (long long)N * 128, 0.f);
    k_edge_scatter_gcn<<<gsz(Etot * 32, 256), B256, 0, stream>>>(ei, E, Etot, dinv, num2, agg, 128);
    k_relu_bias<<<gsz((long long)N * 128, 256), B256, 0, stream>>>(agg, bg, (long long)N * 128, 128);

    // ---------------- Linear ----------------
    {   // out = f3 @ Wl + bl
        dim3 g((N + 15) / 16, (128 + 63) / 64);
        k_gemm_wmma<<<g, B128, 0, stream>>>(agg, Wl, bl, out, N, 128, 128);
    }
}
